// Decoder_20813411516744
// MI455X (gfx1250) — compile-verified, matrix operand order checked by reading, not run on hardware
//
#include <hip/hip_runtime.h>
#include <hip/hip_bf16.h>
#include <math.h>

typedef __attribute__((ext_vector_type(16))) __bf16 v16bf;
typedef __attribute__((ext_vector_type(8)))  float  v8f;

#define B_ 256
#define R_ 196
#define E_ 512
#define D_ 512
#define V_ 10000
#define T_ 26
#define TS 25   // T-1 steps

#define TM 128
#define TN 64
#define TK 32

__device__ __forceinline__ float fast_tanh(float x){
#if __has_builtin(__builtin_amdgcn_tanhf)
  return __builtin_amdgcn_tanhf(x);
#else
  return tanhf(x);
#endif
}
__device__ __forceinline__ float sigmoidf_(float x){ return 1.0f/(1.0f+__expf(-x)); }

// async global->LDS 16B per lane (CDNA5 ASYNCcnt path), inline asm per ISA 15.18.3
__device__ __forceinline__ void async_b128(unsigned lds_off, const void* g){
  asm volatile("global_load_async_to_lds_b128 %0, %1, off"
               :: "v"(lds_off), "v"((unsigned long long)(size_t)g) : "memory");
}
__device__ __forceinline__ void wait_async0(){
  asm volatile("s_wait_asynccnt 0x0" ::: "memory");
}

// ---------------- generic f32 -> bf16 convert ----------------
__global__ void k_f2bf(const float* __restrict__ in, __bf16* __restrict__ out, int n){
  int i = blockIdx.x*blockDim.x + threadIdx.x;
  if(i < n) out[i] = (__bf16)in[i];
}

// ---------------- build Wcat = [W_ih | W_hh] (bf16) ----------------
__global__ void k_wcat(const float* __restrict__ Wih, const float* __restrict__ Whh,
                       __bf16* __restrict__ wc){
  int i = blockIdx.x*blockDim.x + threadIdx.x;
  if(i >= 2048*1536) return;
  int n = i/1536, k = i%1536;
  float v = (k < 1024) ? Wih[n*1024 + k] : Whh[n*512 + (k-1024)];
  wc[i] = (__bf16)v;
}

__global__ void k_bsum(const float* __restrict__ a, const float* __restrict__ b,
                       float* __restrict__ o, int n){
  int i = blockIdx.x*blockDim.x + threadIdx.x;
  if(i < n) o[i] = a[i] + b[i];
}

// ---------------- mean-pool image features -> bf16 ----------------
__global__ void k_avg(const float* __restrict__ img, __bf16* __restrict__ avgb){
  int i = blockIdx.x*blockDim.x + threadIdx.x;
  if(i >= B_*E_) return;
  int b = i/E_, e = i%E_;
  const float* base = img + (long)b*R_*E_ + e;
  float s = 0.f;
  for(int r=0;r<R_;r++) s += base[(long)r*E_];
  avgb[i] = (__bf16)(s*(1.0f/(float)R_));
}

// ---------------- WMMA bf16 GEMM: C[M,N] = act(A[M,K] * Bw[N,K]^T + bias) ----------------
// 128x64 block tile, 8 waves, each wave 32x32 (4 WMMA accumulators).
// Double-buffered LDS fed by async global->LDS DMA (ASYNCcnt), overlapped with WMMA.
// Requires: K % 32 == 0, M % 128 == 0 (true for all call sites). N handled via clamp+store-guard.
__global__ __launch_bounds__(256) void k_gemm_bf16(
    const __bf16* __restrict__ A, int lda,
    const __bf16* __restrict__ Bw, int ldb,
    const float* __restrict__ bias,
    float* __restrict__ C, __bf16* __restrict__ Cbf, long ldc,
    int M, int N, int K, int act)
{
  __shared__ __align__(16) __bf16 lA[2][TM*TK];
  __shared__ __align__(16) __bf16 lB[2][TN*TK];
  const int tid   = threadIdx.x;
  const int lane  = tid & 31;
  const int wv    = tid >> 5;
  const long mBase = (long)blockIdx.y*TM;
  const long nBase = (long)blockIdx.x*TN;
  const int mwarp = (wv>>1)*32;    // 0,32,64,96
  const int ncol  = (wv&1)*32;     // 0 or 32
  const int l16   = lane & 15;
  const int half8 = (lane>>4)*8;

  v8f acc00={},acc01={},acc10={},acc11={};

  // per-thread 16B staging chunks: A rows arow0 and arow0+64, B row brow
  const int arow0 = tid>>2;          // 0..63
  const int acol  = (tid&3)*8;       // bf16 elems: 0,8,16,24

  // row clamping keeps addresses legal & aligned; garbage only feeds never-stored outputs
  long am0 = mBase + arow0;      if(am0 > M-1) am0 = M-1;
  long am1 = mBase + arow0 + 64; if(am1 > M-1) am1 = M-1;
  long bn  = nBase + arow0;      if(bn  > N-1) bn  = N-1;
  const __bf16* ga0 = A  + am0*lda + acol;
  const __bf16* ga1 = A  + am1*lda + acol;
  const __bf16* gb  = Bw + bn *ldb + acol;

  const int nk = K/TK;
  // prologue: stage 0
  {
    async_b128((unsigned)(size_t)&lA[0][arow0*TK + acol],      ga0);
    async_b128((unsigned)(size_t)&lA[0][(arow0+64)*TK + acol], ga1);
    async_b128((unsigned)(size_t)&lB[0][arow0*TK + acol],      gb);
  }
  wait_async0();
  __syncthreads();

  for(int i=0;i<nk;i++){
    const __bf16* pa = lA[i&1];
    const __bf16* pb = lB[i&1];
    // fragment packing per CDNA5 16-bit layout:
    // VGPR q<4: K=2q,2q+1 (+8 for lanes 16-31); q>=4: K=16+2(q-4) (+8)
    v16bf a0,a1,b0,b1;
    #pragma unroll
    for(int q=0;q<8;q++){
      int kb = ((q<4) ? 2*q : 16 + 2*(q-4)) + half8;
      a0[2*q]   = pa[(mwarp+l16)*TK + kb];     a0[2*q+1] = pa[(mwarp+l16)*TK + kb+1];
      a1[2*q]   = pa[(mwarp+16+l16)*TK + kb];  a1[2*q+1] = pa[(mwarp+16+l16)*TK + kb+1];
      b0[2*q]   = pb[(ncol+l16)*TK + kb];      b0[2*q+1] = pb[(ncol+l16)*TK + kb+1];
      b1[2*q]   = pb[(ncol+16+l16)*TK + kb];   b1[2*q+1] = pb[(ncol+16+l16)*TK + kb+1];
    }
    __syncthreads();   // all waves' fragment reads done -> safe to overwrite other buffer

    const int more = (i+1 < nk);
    if(more){          // DMA next tile while WMMAs run
      int nb = (i+1)&1;
      int ko = (i+1)*TK;
      async_b128((unsigned)(size_t)&lA[nb][arow0*TK + acol],      ga0 + ko);
      async_b128((unsigned)(size_t)&lA[nb][(arow0+64)*TK + acol], ga1 + ko);
      async_b128((unsigned)(size_t)&lB[nb][arow0*TK + acol],      gb  + ko);
    }

    acc00 = __builtin_amdgcn_wmma_f32_16x16x32_bf16(false, a0, false, b0, (short)0, acc00, false, false);
    acc01 = __builtin_amdgcn_wmma_f32_16x16x32_bf16(false, a0, false, b1, (short)0, acc01, false, false);
    acc10 = __builtin_amdgcn_wmma_f32_16x16x32_bf16(false, a1, false, b0, (short)0, acc10, false, false);
    acc11 = __builtin_amdgcn_wmma_f32_16x16x32_bf16(false, a1, false, b1, (short)0, acc11, false, false);

    if(more){
      wait_async0();
      __syncthreads();
    }
  }

  // C/D layout: VGPR j -> M = j + half8 within 16-row group; lane&15 -> N
  #pragma unroll
  for(int j=0;j<8;j++){
    int mo = half8 + j;
    long m0 = mBase + mwarp + mo;
    long m1 = mBase + mwarp + 16 + mo;
    long n0 = nBase + ncol + l16;
    long n1 = nBase + ncol + 16 + l16;
    float vals[4] = {acc00[j], acc01[j], acc10[j], acc11[j]};
    long  ms[4]   = {m0, m0, m1, m1};
    long  ns[4]   = {n0, n1, n0, n1};
    #pragma unroll
    for(int p=0;p<4;p++){
      long m = ms[p], n = ns[p];
      if(m < M && n < N){
        float v = vals[p] + (bias ? bias[n] : 0.0f);
        if(act==1) v = fast_tanh(v); else if(act==2) v = sigmoidf_(v);
        if(C)   C  [m*ldc + n] = v;
        if(Cbf) Cbf[m*ldc + n] = (__bf16)v;
      }
    }
  }
}

// ---------------- attention scores: e[b,r] = sum_d tanh(hU[b,d]+Ws[b,r,d])*v[d] + vb ----------------
__global__ __launch_bounds__(256) void k_score(
    const float* __restrict__ hU, const __bf16* __restrict__ Ws,
    const float* __restrict__ vw, const float* __restrict__ vb,
    float* __restrict__ e)
{
  int idx  = blockIdx.x*8 + (threadIdx.x>>5);   // one wave per (b,r)
  int lane = threadIdx.x & 31;
  int b = idx / R_;
  const float*  hp = hU + (long)b*D_   + lane*16;
  const __bf16* wp = Ws + (long)idx*D_ + lane*16;
  const float*  vp = vw + lane*16;
  float s = 0.f;
  #pragma unroll
  for(int q=0;q<16;q++)
    s += fast_tanh(hp[q] + (float)wp[q]) * vp[q];
  #pragma unroll
  for(int o=16;o>0;o>>=1) s += __shfl_xor(s, o, 32);
  if(lane==0) e[idx] = s + vb[0];
}

// ---------------- softmax over R per batch; also writes alphas output ----------------
__global__ __launch_bounds__(256) void k_softmax(
    const float* __restrict__ e, float* __restrict__ alpha,
    float* __restrict__ out_alpha, long out_stride)
{
  __shared__ float red[256];
  int b = blockIdx.x, tid = threadIdx.x;
  float v = (tid < R_) ? e[(long)b*R_ + tid] : -3.4e38f;
  red[tid] = v; __syncthreads();
  for(int o=128;o>0;o>>=1){ if(tid<o) red[tid] = fmaxf(red[tid], red[tid+o]); __syncthreads(); }
  float mx = red[0]; __syncthreads();
  float ex = (tid < R_) ? __expf(v - mx) : 0.f;
  red[tid] = ex; __syncthreads();
  for(int o=128;o>0;o>>=1){ if(tid<o) red[tid] += red[tid+o]; __syncthreads(); }
  float inv = 1.f/red[0];
  if(tid < R_){
    float a = ex*inv;
    alpha[(long)b*R_ + tid] = a;
    out_alpha[(long)b*out_stride + tid] = a;
  }
}

// ---------------- context[b,e] = sum_r alpha[b,r]*img_bf[b,r,e] ----------------
__global__ __launch_bounds__(256) void k_context(
    const float* __restrict__ alpha, const __bf16* __restrict__ img,
    float* __restrict__ ctx)
{
  __shared__ float sa[R_];
  int b = blockIdx.x, tid = threadIdx.x;
  if(tid < R_) sa[tid] = alpha[(long)b*R_ + tid];
  __syncthreads();
  for(int e0=tid; e0<E_; e0+=256){
    const __bf16* base = img + (long)b*R_*E_ + e0;
    float s = 0.f;
    for(int r=0;r<R_;r++) s += sa[r]*(float)base[(long)r*E_];
    ctx[(long)b*E_ + e0] = s;
  }
}

// ---------------- x_full = [emb[cap_t], gate*ctx, h] in bf16 ----------------
__global__ void k_assemble_x(const int* __restrict__ captions, int t,
    const __bf16* __restrict__ embb, const float* __restrict__ gate,
    const float* __restrict__ ctx, const __bf16* __restrict__ hbf,
    __bf16* __restrict__ x)
{
  int i = blockIdx.x*blockDim.x + threadIdx.x;
  if(i >= B_*1536) return;
  int b = i/1536, j = i%1536;
  __bf16 v;
  if(j < 512){
    int tok = captions[b*T_ + t];
    v = embb[(long)tok*D_ + j];
  } else if(j < 1024){
    int k = j-512;
    v = (__bf16)(gate[b*E_ + k]*ctx[b*E_ + k]);
  } else {
    v = hbf[b*D_ + (j-1024)];
  }
  x[i] = v;
}

// ---------------- LSTM pointwise update ----------------
__global__ void k_lstm(const float* __restrict__ gates,
    float* __restrict__ h, float* __restrict__ c, __bf16* __restrict__ hbf)
{
  int i = blockIdx.x*blockDim.x + threadIdx.x;
  if(i >= B_*D_) return;
  int b = i/D_, d = i%D_;
  const float* g = gates + (long)b*4*D_;
  float ig = sigmoidf_(g[d]);
  float fg = sigmoidf_(g[D_ + d]);
  float gg = fast_tanh(g[2*D_ + d]);
  float og = sigmoidf_(g[3*D_ + d]);
  float cn = fg*c[i] + ig*gg;
  float hn = og*fast_tanh(cn);
  c[i] = cn; h[i] = hn; hbf[i] = (__bf16)hn;
}

extern "C" void kernel_launch(void* const* d_in, const int* in_sizes, int n_in,
                              void* d_out, int out_size, void* d_ws, size_t ws_size,
                              hipStream_t stream) {
  const float* img   = (const float*)d_in[0];
  const int*   caps  = (const int*)  d_in[1];
  const float* U_w   = (const float*)d_in[2],  *U_b  = (const float*)d_in[3];
  const float* W_w   = (const float*)d_in[4],  *W_b  = (const float*)d_in[5];
  const float* v_w   = (const float*)d_in[6],  *v_b  = (const float*)d_in[7];
  const float* ih_w  = (const float*)d_in[8],  *ih_b = (const float*)d_in[9];
  const float* ic_w  = (const float*)d_in[10], *ic_b = (const float*)d_in[11];
  const float* fb_w  = (const float*)d_in[12], *fb_b = (const float*)d_in[13];
  const float* out_w = (const float*)d_in[14], *out_b= (const float*)d_in[15];
  const float* emb   = (const float*)d_in[16];
  const float* W_ih  = (const float*)d_in[17], *W_hh = (const float*)d_in[18];
  const float* b_ih  = (const float*)d_in[19], *b_hh = (const float*)d_in[20];

  float* preds      = (float*)d_out;                     // [B, TS, V]
  float* alphas_out = preds + (size_t)B_*TS*V_;          // [B, TS, R]

  // ---- workspace layout ----
  char* wsb = (char*)d_ws; size_t off = 0;
  auto alloc = [&](size_t bytes)->char*{ char* p = wsb+off; off += (bytes+255)&~(size_t)255; return p; };
  __bf16* img_bf  = (__bf16*)alloc((size_t)B_*R_*E_*2);
  __bf16* Ws_bf   = (__bf16*)alloc((size_t)B_*R_*D_*2);
  __bf16* Uw_bf   = (__bf16*)alloc((size_t)D_*D_*2);
  __bf16* Ww_bf   = (__bf16*)alloc((size_t)D_*E_*2);
  __bf16* fbw_bf  = (__bf16*)alloc((size_t)E_*D_*2);
  __bf16* ihw_bf  = (__bf16*)alloc((size_t)D_*E_*2);
  __bf16* icw_bf  = (__bf16*)alloc((size_t)D_*E_*2);
  __bf16* outw_bf = (__bf16*)alloc((size_t)V_*D_*2);
  __bf16* emb_bf  = (__bf16*)alloc((size_t)V_*D_*2);
  __bf16* wcat_bf = (__bf16*)alloc((size_t)2048*1536*2);
  float*  bsum    = (float*) alloc(2048*4);
  __bf16* avg_bf  = (__bf16*)alloc((size_t)B_*E_*2);
  float*  hbuf    = (float*) alloc((size_t)B_*D_*4);
  float*  cbuf    = (float*) alloc((size_t)B_*D_*4);
  __bf16* h_bf    = (__bf16*)alloc((size_t)B_*D_*2);
  float*  hU      = (float*) alloc((size_t)B_*D_*4);
  float*  ebuf    = (float*) alloc((size_t)B_*R_*4);
  float*  alpha   = (float*) alloc((size_t)B_*R_*4);
  float*  ctx     = (float*) alloc((size_t)B_*E_*4);
  float*  gate    = (float*) alloc((size_t)B_*E_*4);
  __bf16* xbf     = (__bf16*)alloc((size_t)B_*1536*2);
  float*  gatesb  = (float*) alloc((size_t)B_*2048*4);

  auto cv = [&](const float* s, __bf16* d, size_t n){
    k_f2bf<<<(unsigned)((n+255)/256),256,0,stream>>>(s,d,(int)n);
  };
  auto gemm = [&](const __bf16* A,int lda,const __bf16* Bw,int ldb,const float* bias,
                  float* C, __bf16* Cbf, long ldc, int M,int N,int K,int act){
    dim3 g((N+TN-1)/TN,(M+TM-1)/TM);
    k_gemm_bf16<<<g,256,0,stream>>>(A,lda,Bw,ldb,bias,C,Cbf,ldc,M,N,K,act);
  };

  // ---- precompute (once per launch; weights are inputs, cannot persist) ----
  cv(img,   img_bf,  (size_t)B_*R_*E_);
  cv(U_w,   Uw_bf,   (size_t)D_*D_);
  cv(W_w,   Ww_bf,   (size_t)D_*E_);
  cv(fb_w,  fbw_bf,  (size_t)E_*D_);
  cv(ih_w,  ihw_bf,  (size_t)D_*E_);
  cv(ic_w,  icw_bf,  (size_t)D_*E_);
  cv(out_w, outw_bf, (size_t)V_*D_);
  cv(emb,   emb_bf,  (size_t)V_*D_);
  k_wcat<<<(2048*1536+255)/256,256,0,stream>>>(W_ih,W_hh,wcat_bf);
  k_bsum<<<8,256,0,stream>>>(b_ih,b_hh,bsum,2048);
  k_avg <<<(B_*E_+255)/256,256,0,stream>>>(img, avg_bf);

  // h0 = tanh(avg @ ih_w^T + ih_b), c0 = tanh(avg @ ic_w^T + ic_b)
  gemm(avg_bf,E_, ihw_bf,E_, ih_b, hbuf, h_bf,   D_, B_, D_, E_, 1);
  gemm(avg_bf,E_, icw_bf,E_, ic_b, cbuf, nullptr,D_, B_, D_, E_, 1);
  // W_s = img @ W_w^T + W_b  (bf16, stays L2-resident across 25 steps)
  gemm(img_bf,E_, Ww_bf,E_, W_b, nullptr, Ws_bf, D_, B_*R_, D_, E_, 0);

  // ---- 25 sequential decode steps ----
  for(int t=0;t<TS;t++){
    // attention query
    gemm(h_bf,D_, Uw_bf,D_, U_b, hU, nullptr, D_, B_, D_, D_, 0);
    // scores, softmax (also writes alphas output), context
    k_score  <<<(B_*R_)/8,256,0,stream>>>(hU, Ws_bf, v_w, v_b, ebuf);
    k_softmax<<<B_,256,0,stream>>>(ebuf, alpha, alphas_out + (size_t)t*R_, (long)TS*R_);
    k_context<<<B_,256,0,stream>>>(alpha, img_bf, ctx);
    // gate = sigmoid(h @ fb_w^T + fb_b)
    gemm(h_bf,D_, fbw_bf,D_, fb_b, gate, nullptr, E_, B_, E_, D_, 2);
    // x_full = [emb_t, gate*ctx, h]; fused gates GEMM with Wcat
    k_assemble_x<<<(B_*1536+255)/256,256,0,stream>>>(caps, t, emb_bf, gate, ctx, h_bf, xbf);
    gemm(xbf,1536, wcat_bf,1536, bsum, gatesb, nullptr, 2048, B_, 2048, 1536, 0);
    // LSTM pointwise update
    k_lstm<<<(B_*D_+255)/256,256,0,stream>>>(gatesb, hbuf, cbuf, h_bf);
    // vocab projection straight into preds[:, t, :]
    gemm(h_bf,D_, outw_bf,D_, out_b, preds + (size_t)t*V_, nullptr, (long)TS*V_, B_, V_, D_, 0);
  }
}
